// Graphormer_78503412236352
// MI455X (gfx1250) — matrix operand order repaired and checked by reading, not compiled, not run on hardware
//
#include <hip/hip_runtime.h>
#include <hip/hip_bf16.h>

typedef __attribute__((ext_vector_type(16))) __bf16 v16bf;
typedef __attribute__((ext_vector_type(8)))  float  v8f;

// ---------------------------------------------------------------------------
// Fragment loader: two 16-byte LDS chunks -> one 32-byte v16bf fragment.
// ---------------------------------------------------------------------------
__device__ inline v16bf ld_frag(const __bf16* p0, const __bf16* p1) {
  union { uint4 q[2]; v16bf v; } u;
  u.q[0] = *(const uint4*)p0;
  u.q[1] = *(const uint4*)p1;
  return u.v;
}

// ---------------------------------------------------------------------------
// bf16 WMMA GEMM, B pre-transposed ([N][K] row-major), fused epilogue:
//   C[m,n] = alpha * (sum_k A[m,k]*Bt[n,k] + bias[n]) + resid[m,n]  (+ GELU)
// Block: 256 threads = 8 wave32 in 2(M) x 4(N); wave tile 32x64 (2x4 WMMA),
// block tile 64x256, K stepped by 32 through LDS. All LDS traffic is b128.
// Batched via blockIdx.z with (z/Hdim, z%Hdim) stride pairs.
// ---------------------------------------------------------------------------
template<bool GELU>
__global__ __launch_bounds__(256) void gemm_bf16_wmma(
    const __bf16* __restrict__ A,  int lda, long sAb, long sAh,
    const __bf16* __restrict__ Bt, int ldb, long sBb, long sBh,
    const float*  __restrict__ bias,
    const float*  __restrict__ resid, int ldr, long sRb, long sRh,
    float*        __restrict__ outf,  int ldo, long sOb, long sOh,
    __bf16*       __restrict__ outb,  int ldp, long sPb, long sPh,
    float alpha, int K, int Hdim)
{
  constexpr int AS = 40;              // padded LDS row strides (bf16 units)
  constexpr int BS = 40;
  __shared__ __bf16 As[64  * AS];     // A tile [64][32]
  __shared__ __bf16 Bs[256 * BS];     // B tile [256 n][32 k]

  const int tid  = threadIdx.x;
  const int z    = blockIdx.z;
  const int zb   = z / Hdim, zh = z % Hdim;
  const long offA = (long)zb * sAb + (long)zh * sAh;
  const long offB = (long)zb * sBb + (long)zh * sBh;
  const long m0   = (long)blockIdx.y * 64;
  const long n0   = (long)blockIdx.x * 256;

  const int wid  = tid >> 5, lane = tid & 31;
  const int wm   = wid >> 2, wn   = wid & 3;      // wave grid 2 (M) x 4 (N)
  const int half = lane >> 4, l16 = lane & 15;

  v8f acc[2][4];
  #pragma unroll
  for (int i = 0; i < 2; ++i)
    #pragma unroll
    for (int j = 0; j < 4; ++j)
      #pragma unroll
      for (int e = 0; e < 8; ++e) acc[i][j][e] = 0.0f;

  const int a_row = tid >> 2, a_seg = tid & 3;    // A staging: 256 x uint4

  for (int k0 = 0; k0 < K; k0 += 32) {
    __syncthreads();
    // ---- stage A tile: 64x32 bf16 = 256 uint4 (one per thread) ----
    *(uint4*)&As[a_row * AS + a_seg * 8] =
        *(const uint4*)(A + offA + (m0 + a_row) * (long)lda + k0 + a_seg * 8);

    // ---- stage B tile: 256x32 bf16 = 1024 uint4 (four per thread) ----
    #pragma unroll
    for (int it = 0; it < 4; ++it) {
      int u = tid + it * 256;
      int nr = u >> 2, seg = u & 3;
      *(uint4*)&Bs[nr * BS + seg * 8] =
          *(const uint4*)(Bt + offB + (n0 + nr) * (long)ldb + k0 + seg * 8);
    }
    __syncthreads();

    if (k0 + 32 < K) {                            // global_prefetch_b8 next tiles
      __builtin_prefetch(A  + offA + (m0 + a_row) * (long)lda + (k0 + 32) + a_seg * 8, 0, 3);
      __builtin_prefetch(Bt + offB + (n0 + (tid >> 1)) * (long)ldb + (k0 + 32), 0, 3);
    }

    // ---- fragments (ISA 7.12.2 layouts) + 8 WMMAs per wave ----
    v16bf af[2], bfr[4];
    #pragma unroll
    for (int i = 0; i < 2; ++i) {
      const __bf16* base = &As[(wm * 32 + i * 16 + l16) * AS];
      af[i] = ld_frag(base + half * 8, base + 16 + half * 8);
    }
    #pragma unroll
    for (int j = 0; j < 4; ++j) {
      const __bf16* base = &Bs[(wn * 64 + j * 16 + l16) * BS];
      bfr[j] = ld_frag(base + half * 8, base + 16 + half * 8);
    }
    #pragma unroll
    for (int i = 0; i < 2; ++i)
      #pragma unroll
      for (int j = 0; j < 4; ++j)
        acc[i][j] = __builtin_amdgcn_wmma_f32_16x16x32_bf16(
            false, af[i], false, bfr[j], (short)0, acc[i][j], false, false);
  }

  // ---- epilogue ----
  const long offR = (long)zb * sRb + (long)zh * sRh;
  const long offO = (long)zb * sOb + (long)zh * sOh;
  const long offP = (long)zb * sPb + (long)zh * sPh;
  #pragma unroll
  for (int i = 0; i < 2; ++i)
    #pragma unroll
    for (int j = 0; j < 4; ++j) {
      const long nn = n0 + wn * 64 + j * 16 + l16;
      const float bv = bias ? bias[nn] : 0.0f;
      #pragma unroll
      for (int r = 0; r < 8; ++r) {
        const long mm = m0 + wm * 32 + i * 16 + r + half * 8;
        float v = (acc[i][j][r] + bv) * alpha;
        if (resid) v += resid[offR + mm * ldr + nn];
        if (GELU)  v = 0.5f * v * (1.0f + erff(v * 0.70710678118654752f));
        if (outf)  outf[offO + mm * ldo + nn] = v;
        if (outb)  outb[offP + mm * ldp + nn] = (__bf16)v;
      }
    }
}

// ---------------------------------------------------------------------------
// Tiled transpose (any batch view) with f32/bf16 source -> bf16 dest:
//   dst[z][c][r] = (bf16) src[z][r][c]
// ---------------------------------------------------------------------------
template<typename SrcT>
__global__ __launch_bounds__(256) void transpose_to_bf16(
    const SrcT* __restrict__ src, long sRow, long sB, long sH, int Hdim,
    __bf16* __restrict__ dst, long dRow, long dB, long dH, int R, int C)
{
  __shared__ __bf16 t[32][33];
  const int z = blockIdx.z, zb = z / Hdim, zh = z % Hdim;
  const SrcT* s = src + (long)zb * sB + (long)zh * sH;
  __bf16*    d = dst + (long)zb * dB + (long)zh * dH;
  const int c0 = blockIdx.x * 32, r0 = blockIdx.y * 32;
  const int tx = threadIdx.x & 31, ty = threadIdx.x >> 5;   // 32 x 8
  #pragma unroll
  for (int i = ty; i < 32; i += 8) {
    int r = r0 + i, c = c0 + tx;
    t[i][tx] = (r < R && c < C) ? (__bf16)s[(long)r * sRow + c] : (__bf16)0.0f;
  }
  __syncthreads();
  #pragma unroll
  for (int i = ty; i < 32; i += 8) {
    int c = c0 + i, r = r0 + tx;
    if (c < C && r < R) d[(long)c * dRow + r] = t[tx][i];
  }
}

// ---------------------------------------------------------------------------
// Rank from stable double-argsort(-x): rank_i[j] = #{k: x[k]>x[j]} + ties<j.
// ---------------------------------------------------------------------------
__global__ __launch_bounds__(256) void rank_kernel(const float* __restrict__ x,
                                                   __bf16* __restrict__ out) {
  __shared__ float rowv[1280];
  const long r = blockIdx.x;
  const int tid = threadIdx.x;
  for (int j = tid; j < 1280; j += 256) rowv[j] = x[r * 1280 + j];
  __syncthreads();
  for (int j = tid; j < 1280; j += 256) {
    const float xj = rowv[j];
    int cnt = 0;
    for (int k = 0; k < 1280; ++k) {
      const float xk = rowv[k];
      cnt += (xk > xj) || (xk == xj && k < j);
    }
    out[r * 1280 + j] = (__bf16)((float)(1280 - cnt) * (1.0f / 1280.0f));
  }
}

// ---------------------------------------------------------------------------
__global__ __launch_bounds__(256) void ln_bf16_kernel(
    const float* __restrict__ in, const float* __restrict__ g,
    const float* __restrict__ b, __bf16* __restrict__ out, int L) {
  __shared__ float red[256];
  const long row = blockIdx.x;
  const int tid = threadIdx.x;
  float s = 0.f, s2 = 0.f;
  for (int j = tid; j < L; j += 256) {
    float v = in[row * L + j];
    s += v; s2 += v * v;
  }
  red[tid] = s; __syncthreads();
  for (int o = 128; o > 0; o >>= 1) { if (tid < o) red[tid] += red[tid + o]; __syncthreads(); }
  const float mean = red[0] / L; __syncthreads();
  red[tid] = s2; __syncthreads();
  for (int o = 128; o > 0; o >>= 1) { if (tid < o) red[tid] += red[tid + o]; __syncthreads(); }
  const float var = red[0] / L - mean * mean;
  const float rs  = rsqrtf(var + 1e-5f);
  for (int j = tid; j < L; j += 256)
    out[row * L + j] = (__bf16)((in[row * L + j] - mean) * rs * g[j] + b[j]);
}

// ---------------------------------------------------------------------------
__global__ __launch_bounds__(256) void softmax_bf16_kernel(
    const float* __restrict__ S, __bf16* __restrict__ P) {
  __shared__ float red[256];
  const long row = blockIdx.x;
  const int j = threadIdx.x;
  const float v = S[row * 256 + j];
  red[j] = v; __syncthreads();
  for (int o = 128; o > 0; o >>= 1) { if (j < o) red[j] = fmaxf(red[j], red[j + o]); __syncthreads(); }
  const float mx = red[0]; __syncthreads();
  const float e = expf(v - mx);
  red[j] = e; __syncthreads();
  for (int o = 128; o > 0; o >>= 1) { if (j < o) red[j] += red[j + o]; __syncthreads(); }
  P[row * 256 + j] = (__bf16)(e / red[0]);
}

// ---------------------------------------------------------------------------
// conv_w [co][ci][t] f32 -> wt[t][co][ci] bf16  (A-matrices for conv GEMMs)
__global__ void convw_kernel(const float* __restrict__ w, __bf16* __restrict__ wt) {
  int i = blockIdx.x * 256 + threadIdx.x;
  if (i < 3 * 256 * 256) {
    int t = i / 65536, rem = i % 65536;
    wt[i] = (__bf16)w[(long)rem * 3 + t];
  }
}

// zero padded columns 0 and 1281 of the [8192][1282] bf16 buffer
__global__ void pad_edge_kernel(__bf16* __restrict__ p) {
  int r = blockIdx.x * 256 + threadIdx.x;
  if (r < 8192) {
    p[(long)r * 1282]        = (__bf16)0.0f;
    p[(long)r * 1282 + 1281] = (__bf16)0.0f;
  }
}

// in-place BN(eval) + ReLU on [B,C,L] f32
__global__ void bn_relu_kernel(float* __restrict__ y, const float* __restrict__ g,
                               const float* __restrict__ b, const float* __restrict__ m,
                               const float* __restrict__ v, long n) {
  long i = (long)blockIdx.x * blockDim.x + threadIdx.x;
  if (i < n) {
    int co = (int)((i / 1280) % 256);
    float t = (y[i] - m[co]) * rsqrtf(v[co] + 1e-5f) * g[co] + b[co];
    y[i] = fmaxf(t, 0.0f);
  }
}

// ---------------------------------------------------------------------------
extern "C" void kernel_launch(void* const* d_in, const int* in_sizes, int n_in,
                              void* d_out, int out_size, void* d_ws, size_t ws_size,
                              hipStream_t stream) {
  (void)in_sizes; (void)n_in; (void)out_size; (void)ws_size;
  const float* x    = (const float*)d_in[0];
  const float* re_w = (const float*)d_in[1];
  const float* re_b = (const float*)d_in[2];
  const float* ln1g = (const float*)d_in[3];
  const float* ln1b = (const float*)d_in[4];
  const float* ln2g = (const float*)d_in[5];
  const float* ln2b = (const float*)d_in[6];
  const float* wq = (const float*)d_in[7];   const float* bq = (const float*)d_in[8];
  const float* wk = (const float*)d_in[9];   const float* bk = (const float*)d_in[10];
  const float* wv = (const float*)d_in[11];  const float* bv = (const float*)d_in[12];
  const float* wo = (const float*)d_in[13];  const float* bo = (const float*)d_in[14];
  const float* w1 = (const float*)d_in[15];  const float* b1 = (const float*)d_in[16];
  const float* w2 = (const float*)d_in[17];  const float* b2 = (const float*)d_in[18];
  const float* cw = (const float*)d_in[19];
  const float* bng = (const float*)d_in[20]; const float* bnb = (const float*)d_in[21];
  const float* bnm = (const float*)d_in[22]; const float* bnv = (const float*)d_in[23];

  constexpr long Lh = 1280, Ch = 256, Bh = 32, Hh = 5, Dh = 256;
  constexpr long MR  = Bh * Ch;                 // 8192 rows
  constexpr long WSZ = Lh * Lh;                 // weight elems
  constexpr long SZP = MR * (Lh + 2) * 2;       // padded bf16 slot (21.0 MB)

  // ---- workspace carving (lifetime-aliased; ~212 MB total) ----
  char* p = (char*)d_ws;
  __bf16* wbf   = (__bf16*)p; p += 7 * WSZ * 2;        // W^T (bf16): re,q,k,v,o,w1,w2
  __bf16* wtcv  = (__bf16*)p; p += 3 * 256 * 256 * 2;  // conv taps [t][co][ci]
  float*  nodef = (float*) p; p += MR * Lh * 4;        // node (f32)
  float*  bufF  = (float*) p; p += MR * Lh * 4;        // S, later h (f32)
  __bf16* bf1   = (__bf16*)p; p += SZP;                // rank -> y -> P -> y2
  __bf16* bf2   = (__bf16*)p; p += SZP;                // q -> V^T -> f
  __bf16* bf3   = (__bf16*)p; p += SZP;                // k -> h2 padded
  __bf16* bf4   = (__bf16*)p; p += SZP;                // v -> h2^T
  __bf16* bf5   = (__bf16*)p; p += SZP;                // o

  const dim3 blk(256);
  const dim3 g_big((unsigned)(Lh / 256), (unsigned)(MR / 64), 1);     // 5 x 128
  const dim3 g_att(1, (unsigned)(Ch / 64), (unsigned)(Bh * Hh));      // 1 x 4 x 160
  const dim3 g_cnv((unsigned)(Lh / 256), (unsigned)(Ch / 64), (unsigned)Bh);

  // ---- weight transpose+convert: W[k][n] f32 -> W^T[n][k] bf16 ----
  const float* wsrc[7] = {re_w, wq, wk, wv, wo, w1, w2};
  for (int i = 0; i < 7; ++i)
    transpose_to_bf16<float><<<dim3(40, 40, 1), blk, 0, stream>>>(
        wsrc[i], Lh, 0, 0, 1, wbf + i * WSZ, Lh, 0, 0, (int)Lh, (int)Lh);
  convw_kernel<<<(3 * 256 * 256 + 255) / 256, blk, 0, stream>>>(cw, wtcv);

  // ---- rank ----
  rank_kernel<<<(unsigned)MR, blk, 0, stream>>>(x, bf1);

  // ---- node = x + rank @ re_w + re_b ----
  gemm_bf16_wmma<false><<<g_big, blk, 0, stream>>>(
      bf1, (int)Lh, 0, 0,  wbf + 0 * WSZ, (int)Lh, 0, 0,  re_b,
      x, (int)Lh, 0, 0,  nodef, (int)Lh, 0, 0,  nullptr, 0, 0, 0,
      1.0f, (int)Lh, 1);

  // ---- LN1 -> y (bf16) ----
  ln_bf16_kernel<<<(unsigned)MR, blk, 0, stream>>>(nodef, ln1g, ln1b, bf1, (int)Lh);

  // ---- Q (scaled), K, V projections (bf16 outputs) ----
  gemm_bf16_wmma<false><<<g_big, blk, 0, stream>>>(
      bf1, (int)Lh, 0, 0,  wbf + 1 * WSZ, (int)Lh, 0, 0,  bq,
      nullptr, 0, 0, 0,  nullptr, 0, 0, 0,  bf2, (int)Lh, 0, 0,
      0.0625f /* D^-0.5 */, (int)Lh, 1);
  gemm_bf16_wmma<false><<<g_big, blk, 0, stream>>>(
      bf1, (int)Lh, 0, 0,  wbf + 2 * WSZ, (int)Lh, 0, 0,  bk,
      nullptr, 0, 0, 0,  nullptr, 0, 0, 0,  bf3, (int)Lh, 0, 0,
      1.0f, (int)Lh, 1);
  gemm_bf16_wmma<false><<<g_big, blk, 0, stream>>>(
      bf1, (int)Lh, 0, 0,  wbf + 3 * WSZ, (int)Lh, 0, 0,  bv,
      nullptr, 0, 0, 0,  nullptr, 0, 0, 0,  bf4, (int)Lh, 0, 0,
      1.0f, (int)Lh, 1);

  // ---- S = Q K^T (per b,h); K rows are already [N][K] ----
  gemm_bf16_wmma<false><<<g_att, blk, 0, stream>>>(
      bf2, (int)Lh, Ch * Lh, Dh,   bf3, (int)Lh, Ch * Lh, Dh,   nullptr,
      nullptr, 0, 0, 0,
      bufF, (int)Ch, Hh * Ch * Ch, Ch * Ch,
      nullptr, 0, 0, 0,
      1.0f, (int)Dh, (int)Hh);

  // ---- softmax rows of 256 -> P (bf16); q dead -> transpose V into bf2 ----
  softmax_bf16_kernel<<<(unsigned)(Bh * Hh * Ch), blk, 0, stream>>>(bufF, bf1);
  transpose_to_bf16<__bf16><<<dim3(8, 8, (unsigned)(Bh * Hh)), blk, 0, stream>>>(
      bf4, Lh, Ch * Lh, Dh, (int)Hh,
      bf2, Dh, Hh * Ch * Dh, Ch * Dh, (int)Ch, (int)Dh);

  // ---- O = P V  (B = V^T [d][c]) -> o (bf16, laid back into [B,C,L]) ----
  gemm_bf16_wmma<false><<<g_att, blk, 0, stream>>>(
      bf1, (int)Ch, Hh * Ch * Ch, Ch * Ch,
      bf2, (int)Ch, Hh * Ch * Dh, Ch * Dh,   nullptr,
      nullptr, 0, 0, 0,  nullptr, 0, 0, 0,
      bf5, (int)Lh, Ch * Lh, Dh,
      1.0f, (int)Ch, (int)Hh);

  // ---- h = node + o @ wo + bo (f32) ----
  gemm_bf16_wmma<false><<<g_big, blk, 0, stream>>>(
      bf5, (int)Lh, 0, 0,  wbf + 4 * WSZ, (int)Lh, 0, 0,  bo,
      nodef, (int)Lh, 0, 0,  bufF, (int)Lh, 0, 0,  nullptr, 0, 0, 0,
      1.0f, (int)Lh, 1);

  // ---- LN2 -> y2 (bf16) ----
  ln_bf16_kernel<<<(unsigned)MR, blk, 0, stream>>>(bufF, ln2g, ln2b, bf1, (int)Lh);

  // ---- f = gelu(y2 @ w1 + b1) (bf16) ----
  gemm_bf16_wmma<true><<<g_big, blk, 0, stream>>>(
      bf1, (int)Lh, 0, 0,  wbf + 5 * WSZ, (int)Lh, 0, 0,  b1,
      nullptr, 0, 0, 0,  nullptr, 0, 0, 0,  bf2, (int)Lh, 0, 0,
      1.0f, (int)Lh, 1);

  // ---- h2 = h + f @ w2 + b2, written bf16 into padded conv buffer ----
  pad_edge_kernel<<<(8192 + 255) / 256, blk, 0, stream>>>(bf3);
  gemm_bf16_wmma<false><<<g_big, blk, 0, stream>>>(
      bf2, (int)Lh, 0, 0,  wbf + 6 * WSZ, (int)Lh, 0, 0,  b2,
      bufF, (int)Lh, 0, 0,  nullptr, 0, 0, 0,
      bf3 + 1, (int)(Lh + 2), 0, 0,
      1.0f, (int)Lh, 1);

  // ---- transpose padded h2: [b][ci][p] -> [b][p][ci] (for BT conv GEMMs) ----
  transpose_to_bf16<__bf16><<<dim3(41, 8, (unsigned)Bh), blk, 0, stream>>>(
      bf3, Lh + 2, Ch * (Lh + 2), 0, 1,
      bf4, Ch, (Lh + 2) * Ch, 0, (int)Ch, (int)(Lh + 2));

  // ---- Conv1d(k=3) as 3 accumulated WMMA GEMMs per batch into d_out ----
  float* outp = (float*)d_out;
  for (int t = 0; t < 3; ++t)
    gemm_bf16_wmma<false><<<g_cnv, blk, 0, stream>>>(
        wtcv + (long)t * 65536, 256, 0, 0,
        bf4 + (long)t * 256, (int)Ch, (Lh + 2) * Ch, 0,
        nullptr,
        (t == 0 ? nullptr : outp), (int)Lh, Ch * Lh, 0,
        outp, (int)Lh, Ch * Lh, 0,
        nullptr, 0, 0, 0,
        1.0f, 256, 1);

  // ---- BN(eval) + ReLU in place ----
  const long ntot = Bh * Ch * Lh;
  bn_relu_kernel<<<(unsigned)((ntot + 255) / 256), blk, 0, stream>>>(
      outp, bng, bnb, bnm, bnv, ntot);
}
